// NonAutoRegressive_28209345200826
// MI455X (gfx1250) — matrix operand order repaired and checked by reading, not compiled, not run on hardware
//
#include <hip/hip_runtime.h>
#include <hip/hip_bf16.h>

// ---------------------------------------------------------------------------
// NonAutoRegressive transformer forward for MI455X (gfx1250, wave32, WMMA).
// bf16 WMMA GEMMs with fp32 accumulation, LDS-staged A tiles, flash attention.
// ---------------------------------------------------------------------------

typedef __bf16 bf16_t;
typedef __attribute__((ext_vector_type(16))) __bf16 v16bf;
typedef __attribute__((ext_vector_type(8)))  __bf16 v8bf;
typedef __attribute__((ext_vector_type(8)))  float  v8f;
typedef __attribute__((ext_vector_type(4)))  float  v4f;

// ---- static model config ----
static constexpr int Hd    = 1024;
static constexpr int NH    = 16;
static constexpr int HDh   = 64;      // head dim
static constexpr int DFF   = 4096;
static constexpr int NLAY  = 6;
static constexpr int Cc    = 8;       // codec channels
static constexpr int CV    = 1026;    // codec vocab
static constexpr int Bb    = 2;
static constexpr int TT    = 128;
static constexpr int TE    = 256;
static constexpr int TA    = 512;
static constexpr int IDXMX = 4;
static constexpr int SLEN  = TT + TE + TA + 1;   // 897
static constexpr int NTOK  = Bb * SLEN;          // 1794
static constexpr int SP    = 928;                // key/query padding (29*32)
static constexpr int QTILES = (SLEN + 15) / 16;  // 57

__device__ __forceinline__ v16bf joinbf(v8bf a, v8bf b) {
  v16bf r;
#pragma unroll
  for (int i = 0; i < 8; ++i) { r[i] = a[i]; r[i + 8] = b[i]; }
  return r;
}

__device__ __forceinline__ v8f zero8() {
  v8f z;
#pragma unroll
  for (int i = 0; i < 8; ++i) z[i] = 0.f;
  return z;
}

// ---------------------------------------------------------------------------
// Embedding assembly: x[b,pos,:] per segment + sinusoid, gathered by lengths.
// Writes fp32 master activation x and bf16 "memory" mirror (cross-attn K/V src).
// ---------------------------------------------------------------------------
__device__ __forceinline__ float pe_val(int t, int j) {
  float freq = __expf(-logf(10000.f) * (float)(j & ~1) * (1.0f / (float)Hd));
  float a = (float)t * freq;
  return (j & 1) ? cosf(a) : sinf(a);
}

__global__ void embed_kernel(const int* __restrict__ text, const int* __restrict__ audio,
                             const int* __restrict__ enr,
                             const int* __restrict__ tl_, const int* __restrict__ al_,
                             const int* __restrict__ el_, const int* __restrict__ idxp,
                             const float* __restrict__ text_emb,
                             const float* __restrict__ shared_audio,
                             const float* __restrict__ index_emb,
                             float* __restrict__ x, bf16_t* __restrict__ memb) {
  int row = blockIdx.x;
  if (row >= NTOK) return;
  int idx = *idxp; if (idx > IDXMX) idx = IDXMX;
  int b = row / SLEN, pos = row - b * SLEN;
  int tl = tl_[b], el = el_[b], al = al_[b];
  int s2 = tl + el, s3 = s2 + al;

  for (int j = threadIdx.x; j < Hd; j += 256) {
    float v;
    if (pos < tl) {
      int t = pos;
      v = text_emb[(size_t)text[b * TT + t] * Hd + j] + pe_val(t, j);
    } else if (pos < s2) {
      int t = pos - tl;
      float s = 0.f;
#pragma unroll
      for (int c = 0; c < Cc; ++c)
        s += shared_audio[(size_t)(enr[(b * Cc + c) * TE + t] + c * CV) * Hd + j];
      v = s + pe_val(t, j);
    } else if (pos < s3) {
      int t = pos - s2;
      float s = 0.f;
      for (int c = 0; c < idx; ++c)
        s += shared_audio[(size_t)(audio[(b * IDXMX + c) * TA + t] + c * CV) * Hd + j];
      v = s + pe_val(t, j);
    } else if (pos == s3) {
      v = index_emb[(size_t)idx * Hd + j];
    } else {
      v = 0.f;
    }
    x[(size_t)row * Hd + j] = v;
    memb[(size_t)row * Hd + j] = (bf16_t)v;
  }
}

// ---------------------------------------------------------------------------
// LayerNorm: fp32 in, bf16 out.  Gamma/beta selected by runtime `index`.
// One 256-thread block per token.
// ---------------------------------------------------------------------------
__global__ void ln_kernel(const float* __restrict__ x, const float* __restrict__ wb,
                          const float* __restrict__ bbias, const int* __restrict__ idxp,
                          bf16_t* __restrict__ out) {
  int row = blockIdx.x;
  const float* xr = x + (size_t)row * Hd;
  const float* w = wb + (size_t)(*idxp) * Hd;
  const float* bgt = bbias + (size_t)(*idxp) * Hd;

  float s = 0.f, s2 = 0.f;
  for (int j = threadIdx.x; j < Hd; j += 256) { float v = xr[j]; s += v; s2 += v * v; }
#pragma unroll
  for (int off = 16; off > 0; off >>= 1) { s += __shfl_xor(s, off, 32); s2 += __shfl_xor(s2, off, 32); }
  __shared__ float red[18];
  int wv = threadIdx.x >> 5, lane = threadIdx.x & 31;
  if (lane == 0) { red[wv] = s; red[8 + wv] = s2; }
  __syncthreads();
  if (threadIdx.x == 0) {
    float a = 0.f, c = 0.f;
#pragma unroll
    for (int i = 0; i < 8; ++i) { a += red[i]; c += red[8 + i]; }
    float mean = a * (1.0f / Hd);
    red[16] = mean;
    red[17] = c * (1.0f / Hd) - mean * mean;
  }
  __syncthreads();
  float mean = red[16];
  float rstd = rsqrtf(red[17] + 1e-5f);
  for (int j = threadIdx.x; j < Hd; j += 256) {
    float v = (xr[j] - mean) * rstd * w[j] + bgt[j];
    out[(size_t)row * Hd + j] = (bf16_t)v;
  }
}

// ---------------------------------------------------------------------------
// NT GEMM: C[m,n] = sum_k A_bf16[m,k] * W_f32[n,k]  (+bias, relu, +residual)
// Block = 128 thr (4 waves).  Block tile 64(M) x 64(N); wave w -> 16 N cols,
// 4 stacked 16x16 WMMA accumulators sharing one B fragment per K-step.
// A tile (64x32 bf16) is staged in LDS once per K-step and shared by all
// 4 waves (padded stride 40 -> conflict-free ds_load_b128 a-frag reads).
// Weight stream is prefetched 2 K-steps ahead (global_prefetch_b8).
// Optional epilogues: fp32 out, bf16 out, scatter to [B,NH,SP,64] bf16.
// ---------------------------------------------------------------------------
static constexpr int ATS = 40;  // LDS A-tile row stride (bf16 elems), 16B aligned

__global__ void gemm_nt(const bf16_t* __restrict__ A, const float* __restrict__ Wf,
                        const float* __restrict__ bias, const float* __restrict__ residual,
                        float* __restrict__ outF, bf16_t* __restrict__ outB,
                        bf16_t* __restrict__ outScat,
                        const int* __restrict__ wselp, unsigned long long wstride,
                        int M, int N, int K, int relu) {
  __shared__ __align__(16) bf16_t At[64 * ATS];

  int lane = threadIdx.x & 31, wave = threadIdx.x >> 5;
  int r = lane & 15, hh = lane >> 4;
  int m0 = blockIdx.x * 64;
  int n0 = blockIdx.y * 64 + wave * 16;

  const float* W = Wf + (wselp ? (size_t)(*wselp) * wstride : 0);
  int bn = n0 + r; if (bn >= N) bn = N - 1;                 // clamp weight row
  const float* wp = W + (size_t)bn * K + 16 * hh;           // B-frag: K=16*half+j

  // cooperative A-stage source: thread t copies half-row (16 bf16 = 32B)
  int srow = threadIdx.x >> 1, sseg = threadIdx.x & 1;
  int am = m0 + srow; if (am >= M) am = M - 1;
  const bf16_t* asrc = A + (size_t)am * K + sseg * 16;
  bf16_t* adst = At + srow * ATS + sseg * 16;

  v8f acc[4];
#pragma unroll
  for (int t = 0; t < 4; ++t) acc[t] = zero8();

  for (int k = 0; k < K; k += 32) {
    __syncthreads();
    {  // stage 64x32 A tile into LDS (each thread: two b128 copies)
      v8bf s0 = *(const v8bf*)(asrc + k);
      v8bf s1 = *(const v8bf*)(asrc + k + 8);
      *(v8bf*)(adst) = s0;
      *(v8bf*)(adst + 8) = s1;
    }
    __builtin_prefetch((const void*)(wp + k + 64), 0, 1);   // weight stream ahead
    __syncthreads();

    // b-frag: 16 contiguous fp32 weights -> bf16
    v16bf bfrag;
    const v4f* w4 = (const v4f*)(wp + k);
#pragma unroll
    for (int q4 = 0; q4 < 4; ++q4) {
      v4f wv = w4[q4];
#pragma unroll
      for (int e = 0; e < 4; ++e) bfrag[q4 * 4 + e] = (bf16_t)wv[e];
    }
#pragma unroll
    for (int t = 0; t < 4; ++t) {
      const bf16_t* pa = At + (16 * t + r) * ATS + 16 * hh;  // A-frag K=16*half+j
      v16bf afrag = joinbf(*(const v8bf*)(pa), *(const v8bf*)(pa + 8));
      acc[t] = __builtin_amdgcn_wmma_f32_16x16x32_bf16(false, afrag, false, bfrag,
                                                       (short)0, acc[t], false, false);
    }
  }

  int n = n0 + r;
  if (n < N) {
    float bv = bias ? bias[n] : 0.f;
#pragma unroll
    for (int t = 0; t < 4; ++t) {
#pragma unroll
      for (int i = 0; i < 8; ++i) {
        int m = m0 + 16 * t + i + 8 * hh;                   // C layout: M=i+8*half, N=r
        if (m < M) {
          float v = acc[t][i] + bv;
          if (relu) v = fmaxf(v, 0.f);
          if (residual) v += residual[(size_t)m * N + n];
          if (outF) outF[(size_t)m * N + n] = v;
          if (outB) outB[(size_t)m * N + n] = (bf16_t)v;
          if (outScat) {
            int bat = m / SLEN, s = m - bat * SLEN;
            outScat[(((size_t)(bat * NH + (n >> 6)) * SP) + s) * HDh + (n & 63)] = (bf16_t)v;
          }
        }
      }
    }
  }
}

// ---------------------------------------------------------------------------
// Fused flash attention.  Q/K/V: bf16 [B, NH, SP, 64].
// Grid: (ceil(QTILES/4), B*NH).  Block 128 thr = 4 waves; each wave owns one
// 16-query tile, all waves share K/V LDS tiles per 32-key step.
// QK^T and P*V both via v_wmma_f32_16x16x32_bf16; online softmax across
// half-wave lanes.  Output written as bf16 [NTOK, 1024] (o-proj GEMM input).
// ---------------------------------------------------------------------------
static constexpr int KTS = 72;  // Kt row stride (bf16) -> conflict-free b-frag reads
static constexpr int VTS = 40;  // Vt^T row stride
static constexpr int PTS = 40;  // Pt row stride

__global__ void attn_kernel(const bf16_t* __restrict__ Q, const bf16_t* __restrict__ Kg,
                            const bf16_t* __restrict__ Vg, bf16_t* __restrict__ outBf) {
  __shared__ __align__(16) bf16_t Kt[32 * KTS];
  __shared__ __align__(16) bf16_t Vt[64 * VTS];       // transposed: Vt[d][k]
  __shared__ __align__(16) bf16_t Pt[4][16 * PTS];

  int lane = threadIdx.x & 31, wave = threadIdx.x >> 5;
  int r = lane & 15, hh = lane >> 4;
  int bh = blockIdx.y;
  int b = bh >> 4, head = bh & 15;
  const bf16_t* Qh = Q + (size_t)bh * SP * HDh;
  const bf16_t* Kh = Kg + (size_t)bh * SP * HDh;
  const bf16_t* Vh = Vg + (size_t)bh * SP * HDh;

  int qtile = blockIdx.x * 4 + wave;
  bool qvalid = qtile < QTILES;
  int qbase = qvalid ? qtile * 16 : 0;

  // Q a-frags (row = qbase + r), K = 16*half+j then +32
  const bf16_t* qp = Qh + (size_t)(qbase + r) * HDh;
  v16bf aq0 = joinbf(*(const v8bf*)(qp + 16 * hh), *(const v8bf*)(qp + 8 + 16 * hh));
  v16bf aq1 = joinbf(*(const v8bf*)(qp + 32 + 16 * hh), *(const v8bf*)(qp + 40 + 16 * hh));

  float mrun[8], lrun[8];
  v8f acc[4];
#pragma unroll
  for (int i = 0; i < 8; ++i) { mrun[i] = -1e30f; lrun[i] = 0.f; }
#pragma unroll
  for (int c = 0; c < 4; ++c) acc[c] = zero8();

  const float sc = 0.125f;  // 1/sqrt(64)

  for (int kb = 0; kb < SP; kb += 32) {
    __syncthreads();
    {   // cooperative stage: Kt row-major (padded), Vt transposed (padded)
      int t = threadIdx.x;
      int row = t >> 2, seg = t & 3;
      const bf16_t* ks = Kh + (size_t)(kb + row) * HDh + seg * 16;
      const bf16_t* vs = Vh + (size_t)(kb + row) * HDh + seg * 16;
      bf16_t* kd = Kt + row * KTS + seg * 16;
#pragma unroll
      for (int j = 0; j < 16; ++j) kd[j] = ks[j];
#pragma unroll
      for (int j = 0; j < 16; ++j) Vt[(seg * 16 + j) * VTS + row] = vs[j];
    }
    __syncthreads();

    // ---- scores: two 16-key subtiles, K(contraction)=64 -> 2 WMMAs each ----
    v8f c0 = zero8(), c1 = zero8();
    {
      const bf16_t* kp0 = Kt + r * KTS;            // key = kb + r
      const bf16_t* kp1 = Kt + (16 + r) * KTS;     // key = kb + 16 + r
      v16bf bk;
      bk = joinbf(*(const v8bf*)(kp0 + 16 * hh), *(const v8bf*)(kp0 + 16 * hh + 8));
      c0 = __builtin_amdgcn_wmma_f32_16x16x32_bf16(false, aq0, false, bk, (short)0, c0, false, false);
      bk = joinbf(*(const v8bf*)(kp0 + 32 + 16 * hh), *(const v8bf*)(kp0 + 32 + 16 * hh + 8));
      c0 = __builtin_amdgcn_wmma_f32_16x16x32_bf16(false, aq1, false, bk, (short)0, c0, false, false);
      bk = joinbf(*(const v8bf*)(kp1 + 16 * hh), *(const v8bf*)(kp1 + 16 * hh + 8));
      c1 = __builtin_amdgcn_wmma_f32_16x16x32_bf16(false, aq0, false, bk, (short)0, c1, false, false);
      bk = joinbf(*(const v8bf*)(kp1 + 32 + 16 * hh), *(const v8bf*)(kp1 + 32 + 16 * hh + 8));
      c1 = __builtin_amdgcn_wmma_f32_16x16x32_bf16(false, aq1, false, bk, (short)0, c1, false, false);
    }

    // ---- online softmax (rows live across 16 lanes of each half-wave) ----
    int key0 = kb + r, key1 = kb + 16 + r;
#pragma unroll
    for (int i = 0; i < 8; ++i) {
      float s0 = (key0 < SLEN) ? c0[i] * sc : -1e30f;
      float s1 = (key1 < SLEN) ? c1[i] * sc : -1e30f;
      float t = fmaxf(s0, s1);
#pragma unroll
      for (int off = 1; off < 16; off <<= 1) t = fmaxf(t, __shfl_xor(t, off, 16));
      float mn = fmaxf(mrun[i], t);
      float corr = __expf(mrun[i] - mn);
      mrun[i] = mn;
      float p0 = __expf(s0 - mn);
      float p1 = __expf(s1 - mn);
      float rs = p0 + p1;
#pragma unroll
      for (int off = 1; off < 16; off <<= 1) rs += __shfl_xor(rs, off, 16);
      lrun[i] = lrun[i] * corr + rs;
#pragma unroll
      for (int c = 0; c < 4; ++c) acc[c][i] *= corr;
      Pt[wave][(i + 8 * hh) * PTS + r] = (bf16_t)p0;
      Pt[wave][(i + 8 * hh) * PTS + 16 + r] = (bf16_t)p1;
    }
    __syncthreads();

    // ---- P(16x32) @ V(32x64): 4 WMMAs (one per 16-dim output chunk) ----
    const bf16_t* pp = Pt[wave] + r * PTS;
    v16bf apv = joinbf(*(const v8bf*)(pp + 16 * hh), *(const v8bf*)(pp + 8 + 16 * hh));
#pragma unroll
    for (int c = 0; c < 4; ++c) {
      const bf16_t* vp = Vt + (16 * c + r) * VTS + 16 * hh;
      v16bf bv = joinbf(*(const v8bf*)(vp), *(const v8bf*)(vp + 8));
      acc[c] = __builtin_amdgcn_wmma_f32_16x16x32_bf16(false, apv, false, bv, (short)0, acc[c], false, false);
    }
  }

  if (qvalid) {
#pragma unroll
    for (int i = 0; i < 8; ++i) {
      int q = qbase + i + 8 * hh;
      if (q < SLEN) {
        float inv = 1.f / lrun[i];
        size_t rowg = (size_t)(b * SLEN + q) * Hd + head * HDh;
#pragma unroll
        for (int c = 0; c < 4; ++c)
          outBf[rowg + 16 * c + r] = (bf16_t)(acc[c][i] * inv);
      }
    }
  }
}

// ---------------------------------------------------------------------------
// Host-side orchestration
// ---------------------------------------------------------------------------
static inline void launch_gemm(hipStream_t st, const bf16_t* A, const float* W,
                               const float* bias, const float* res, float* outF,
                               bf16_t* outB, bf16_t* outScat, const int* wsel,
                               unsigned long long wstride, int M, int N, int K, int relu) {
  dim3 g((M + 63) / 64, (N + 63) / 64), blk(128);
  gemm_nt<<<g, blk, 0, st>>>(A, W, bias, res, outF, outB, outScat, wsel, wstride, M, N, K, relu);
}

extern "C" void kernel_launch(void* const* d_in, const int* in_sizes, int n_in,
                              void* d_out, int out_size, void* d_ws, size_t ws_size,
                              hipStream_t stream) {
  (void)in_sizes; (void)n_in; (void)out_size; (void)ws_size;

  // ---- raw inputs (setup_inputs order) ----
  const int* text = (const int*)d_in[0];
  const int* audio = (const int*)d_in[1];
  const int* enr = (const int*)d_in[2];
  const int* tl = (const int*)d_in[3];
  const int* al = (const int*)d_in[4];
  const int* el = (const int*)d_in[5];
  const int* idxp = (const int*)d_in[6];

  // ---- params pytree leaves, sorted-key order ----
  int p = 7;
  const float* index_emb = (const float*)d_in[p++];  // (8,1024)
  const float* b1   = (const float*)d_in[p++];       // (6,4096)
  const float* b2   = (const float*)d_in[p++];       // (6,1024)
  const float* ca_k = (const float*)d_in[p++];       // (6,1024,1024)
  const float* ca_kb= (const float*)d_in[p++];
  const float* ca_o = (const float*)d_in[p++];
  const float* ca_ob= (const float*)d_in[p++];
  const float* ca_q = (const float*)d_in[p++];
  const float* ca_qb= (const float*)d_in[p++];
  const float* ca_v = (const float*)d_in[p++];
  const float* ca_vb= (const float*)d_in[p++];
  const float* ln1_b= (const float*)d_in[p++];       // (6,8,1024)
  const float* ln1_w= (const float*)d_in[p++];
  const float* ln2_b= (const float*)d_in[p++];
  const float* ln2_w= (const float*)d_in[p++];
  const float* ln3_b= (const float*)d_in[p++];
  const float* ln3_w= (const float*)d_in[p++];
  const float* sa_k = (const float*)d_in[p++];
  const float* sa_kb= (const float*)d_in[p++];
  const float* sa_o = (const float*)d_in[p++];
  const float* sa_ob= (const float*)d_in[p++];
  const float* sa_q = (const float*)d_in[p++];
  const float* sa_qb= (const float*)d_in[p++];
  const float* sa_v = (const float*)d_in[p++];
  const float* sa_vb= (const float*)d_in[p++];
  const float* w1   = (const float*)d_in[p++];       // (6,4096,1024)
  const float* w2   = (const float*)d_in[p++];       // (6,1024,4096)
  const float* shared_audio = (const float*)d_in[p++]; // (8,1026,1024)
  const float* text_emb     = (const float*)d_in[p++]; // (256,1024)

  // ---- workspace carve ----
  size_t off = 0;
  auto carve = [&](size_t bytes) {
    void* ptr = (char*)d_ws + off;
    off += (bytes + 255) & ~(size_t)255;
    return ptr;
  };
  float*  x    = (float*)carve((size_t)NTOK * Hd * 4);
  bf16_t* memb = (bf16_t*)carve((size_t)NTOK * Hd * 2);
  bf16_t* hb   = (bf16_t*)carve((size_t)NTOK * Hd * 2);
  bf16_t* aob  = (bf16_t*)carve((size_t)NTOK * Hd * 2);
  bf16_t* xb   = (bf16_t*)carve((size_t)NTOK * Hd * 2);
  bf16_t* qbuf = (bf16_t*)carve((size_t)Bb * NH * SP * HDh * 2);
  bf16_t* kbuf = (bf16_t*)carve((size_t)Bb * NH * SP * HDh * 2);
  bf16_t* vbuf = (bf16_t*)carve((size_t)Bb * NH * SP * HDh * 2);
  bf16_t* ffnb = (bf16_t*)carve((size_t)NTOK * DFF * 2);

  const unsigned long long HH = (unsigned long long)Hd * Hd;
  const unsigned long long FH = (unsigned long long)DFF * Hd;
  const int LNS = Cc * Hd;

  // ---- embedding ----
  embed_kernel<<<dim3(NTOK), dim3(256), 0, stream>>>(
      text, audio, enr, tl, al, el, idxp, text_emb, shared_audio, index_emb, x, memb);

  dim3 attnGrid((QTILES + 3) / 4, Bb * NH), attnBlk(128);

  for (int l = 0; l < NLAY; ++l) {
    // ---- self attention ----
    ln_kernel<<<dim3(NTOK), dim3(256), 0, stream>>>(x, ln1_w + (size_t)l * LNS,
                                                    ln1_b + (size_t)l * LNS, idxp, hb);
    launch_gemm(stream, hb, sa_q + l * HH, sa_qb + l * Hd, nullptr, nullptr, nullptr, qbuf,
                nullptr, 0, NTOK, Hd, Hd, 0);
    launch_gemm(stream, hb, sa_k + l * HH, sa_kb + l * Hd, nullptr, nullptr, nullptr, kbuf,
                nullptr, 0, NTOK, Hd, Hd, 0);
    launch_gemm(stream, hb, sa_v + l * HH, sa_vb + l * Hd, nullptr, nullptr, nullptr, vbuf,
                nullptr, 0, NTOK, Hd, Hd, 0);
    attn_kernel<<<attnGrid, attnBlk, 0, stream>>>(qbuf, kbuf, vbuf, aob);
    launch_gemm(stream, aob, sa_o + l * HH, sa_ob + l * Hd, x, x, nullptr, nullptr,
                nullptr, 0, NTOK, Hd, Hd, 0);

    // ---- cross attention (memory = embed) ----
    ln_kernel<<<dim3(NTOK), dim3(256), 0, stream>>>(x, ln2_w + (size_t)l * LNS,
                                                    ln2_b + (size_t)l * LNS, idxp, hb);
    launch_gemm(stream, hb,   ca_q + l * HH, ca_qb + l * Hd, nullptr, nullptr, nullptr, qbuf,
                nullptr, 0, NTOK, Hd, Hd, 0);
    launch_gemm(stream, memb, ca_k + l * HH, ca_kb + l * Hd, nullptr, nullptr, nullptr, kbuf,
                nullptr, 0, NTOK, Hd, Hd, 0);
    launch_gemm(stream, memb, ca_v + l * HH, ca_vb + l * Hd, nullptr, nullptr, nullptr, vbuf,
                nullptr, 0, NTOK, Hd, Hd, 0);
    attn_kernel<<<attnGrid, attnBlk, 0, stream>>>(qbuf, kbuf, vbuf, aob);
    launch_gemm(stream, aob, ca_o + l * HH, ca_ob + l * Hd, x, x, nullptr, nullptr,
                nullptr, 0, NTOK, Hd, Hd, 0);

    // ---- FFN ----
    ln_kernel<<<dim3(NTOK), dim3(256), 0, stream>>>(x, ln3_w + (size_t)l * LNS,
                                                    ln3_b + (size_t)l * LNS, idxp, hb);
    launch_gemm(stream, hb, w1 + l * FH, b1 + (size_t)l * DFF, nullptr, nullptr, ffnb, nullptr,
                nullptr, 0, NTOK, DFF, Hd, 1);
    // last layer also emits bf16 x mirror for the final projection
    launch_gemm(stream, ffnb, w2 + l * FH, b2 + (size_t)l * Hd, x, x,
                (l == NLAY - 1) ? xb : nullptr, nullptr, nullptr, 0, NTOK, Hd, DFF, 0);
  }

  // ---- final logits: out[b,l,d] = sum_c x[b,l,c] * shared_audio[index][d,c] ----
  launch_gemm(stream, xb, shared_audio, nullptr, nullptr, (float*)d_out, nullptr, nullptr,
              idxp, (unsigned long long)CV * Hd, NTOK, CV, Hd, 0);
}
